// NezNet_46248207843927
// MI455X (gfx1250) — matrix operand order repaired and checked by reading, not compiled, not run on hardware
//
#include <hip/hip_runtime.h>
#include <hip/hip_bf16.h>

// ---------------------------------------------------------------------------
// Problem constants (match reference)
// ---------------------------------------------------------------------------
constexpr int NN = 50000;     // nodes
constexpr int EE = 1600000;   // edges
constexpr int FF = 64;        // input features
constexpr int HH = 128;       // hidden
constexpr int GG = 512;       // graphs
constexpr float BN_EPS = 1e-3f;
constexpr int ROW_TILES = NN / 16;   // 3125 exact

typedef __bf16 bf16_t;
typedef __attribute__((ext_vector_type(16))) __bf16 v16bf;
typedef __attribute__((ext_vector_type(8)))  __bf16 v8bf;
typedef __attribute__((ext_vector_type(8)))  float   v8f;
typedef __attribute__((ext_vector_type(4)))  float   v4f;

// ---------------------------------------------------------------------------
// WMMA helpers (CDNA5 wave32 layouts, 16x16x32 bf16 -> f32)
// A 16x32 bf16: lane l holds row M=l&15; hi=l>>4 selects K-halves.
//   vector element i  <->  K = 8*hi + (i<8 ? i : i+8)   (within 32-wide K block)
// B 32x16 bf16 (from transposed weights Wt[N][K]): lane l col N=l&15;
//   vector element i  <->  K = 8*hi + (i<8 ? i : i+8)   (same order as A)
// C/D 16x16 f32: element r of v8f <-> D[M = r + 8*hi][N = l&15]
// ---------------------------------------------------------------------------
__device__ __forceinline__ v8f wmma_bf16(v16bf a, v16bf b, v8f c) {
  return __builtin_amdgcn_wmma_f32_16x16x32_bf16(
      /*neg_a=*/false, a, /*neg_b=*/false, b,
      /*c_mod=*/(short)0, c, /*reuse_a=*/false, /*reuse_b=*/false);
}

// A fragment from bf16 row-major activations (row base per-lane), K block at k0
__device__ __forceinline__ v16bf load_a_bf16(const bf16_t* __restrict__ rowbase,
                                             int k0, int hi) {
  const bf16_t* p = rowbase + k0 + 8 * hi;
  v8bf lo = *(const v8bf*)(p);
  v8bf h8 = *(const v8bf*)(p + 16);
  v16bf a;
#pragma unroll
  for (int i = 0; i < 8; ++i) { a[i] = lo[i]; a[8 + i] = h8[i]; }
  return a;
}

// A fragment from f32 row-major activations, converted to bf16 on the fly
__device__ __forceinline__ v16bf load_a_f32(const float* __restrict__ rowbase,
                                            int k0, int hi) {
  const float* p = rowbase + k0 + 8 * hi;
  v16bf a;
#pragma unroll
  for (int i = 0; i < 8; ++i) {
    a[i]     = (bf16_t)p[i];
    a[8 + i] = (bf16_t)p[16 + i];
  }
  return a;
}

// B fragment from transposed bf16 weights Wt[N][K] (ldk = K)
__device__ __forceinline__ v16bf load_b_t(const bf16_t* __restrict__ wt, int ldk,
                                          int n0, int k0, int r, int hi) {
  const bf16_t* p = wt + (size_t)(n0 + r) * ldk + k0 + 8 * hi;
  v8bf x0 = *(const v8bf*)(p);
  v8bf x1 = *(const v8bf*)(p + 16);
  v16bf b;
#pragma unroll
  for (int i = 0; i < 8; ++i) { b[i] = x0[i]; b[8 + i] = x1[i]; }
  return b;
}

// ---------------------------------------------------------------------------
// K0: prepare bf16 transposed weights.  wA = W1-W2, wB = W2 (EdgeConv algebra)
// ---------------------------------------------------------------------------
__global__ void k_prep_w(const float* __restrict__ w_pre,
                         const float* __restrict__ w_cn,
                         const float* __restrict__ w_post,
                         bf16_t* __restrict__ wpre_t,   // [HH][FF]
                         bf16_t* __restrict__ wA_t,     // [HH][HH]
                         bf16_t* __restrict__ wB_t,     // [HH][HH]
                         bf16_t* __restrict__ wpost_t)  // [HH][HH]
{
  int tid = blockIdx.x * blockDim.x + threadIdx.x;
  if (tid >= HH * HH) return;
  int k = tid >> 7;       // 0..127
  int c = tid & 127;      // 0..127
  float w1 = w_cn[k * HH + c];
  float w2 = w_cn[(k + HH) * HH + c];
  wA_t[c * HH + k] = (bf16_t)(w1 - w2);
  wB_t[c * HH + k] = (bf16_t)(w2);
  wpost_t[c * HH + k] = (bf16_t)(w_post[k * HH + c]);
  if (k < FF) wpre_t[c * FF + k] = (bf16_t)(w_pre[k * HH + c]);
}

// ---------------------------------------------------------------------------
// K1: y = x @ w_pre + b_pre   ([N,64] @ [64,128]), WMMA, f32 output (pre-BN)
// 8 waves/block, one 16-row tile per wave.
// ---------------------------------------------------------------------------
__global__ void k_gemm_pre(const float* __restrict__ x,
                           const bf16_t* __restrict__ wpre_t,
                           const float* __restrict__ bias,
                           float* __restrict__ y) {
  const int wave = threadIdx.x >> 5;
  const int lane = threadIdx.x & 31;
  const int tile = blockIdx.x * 8 + wave;
  if (tile >= ROW_TILES) return;
  const int r = lane & 15, hi = lane >> 4;
  const int row0 = tile * 16;
  const float* arow = x + (size_t)(row0 + r) * FF;
  v16bf a0 = load_a_f32(arow, 0, hi);
  v16bf a1 = load_a_f32(arow, 32, hi);
#pragma unroll
  for (int nt = 0; nt < 8; ++nt) {
    const int n0 = nt * 16;
    v8f acc = {};
    v16bf b0 = load_b_t(wpre_t, FF, n0, 0, r, hi);
    v16bf b1 = load_b_t(wpre_t, FF, n0, 32, r, hi);
    acc = wmma_bf16(a0, b0, acc);
    acc = wmma_bf16(a1, b1, acc);
    const float bv = bias[n0 + r];
    float* yb = y + (size_t)row0 * HH + n0 + r;
#pragma unroll
    for (int rr = 0; rr < 8; ++rr)
      yb[(size_t)(rr + 8 * hi) * HH] = acc[rr] + bv;
  }
}

// ---------------------------------------------------------------------------
// K2: per-column batch statistics (training-mode BN): mean & rsqrt(var+eps)
// ---------------------------------------------------------------------------
__global__ void k_colstats(const float* __restrict__ m, int rows,
                           float* __restrict__ mean, float* __restrict__ rstd) {
  const int c = blockIdx.x;  // 0..127
  float s = 0.f, s2 = 0.f;
  for (int row = threadIdx.x; row < rows; row += blockDim.x) {
    float v = m[(size_t)row * HH + c];
    s += v;
    s2 += v * v;
  }
  __shared__ float sh[256], sh2[256];
  sh[threadIdx.x] = s;
  sh2[threadIdx.x] = s2;
  __syncthreads();
  for (int o = 128; o > 0; o >>= 1) {
    if ((int)threadIdx.x < o) {
      sh[threadIdx.x] += sh[threadIdx.x + o];
      sh2[threadIdx.x] += sh2[threadIdx.x + o];
    }
    __syncthreads();
  }
  if (threadIdx.x == 0) {
    float inv = 1.f / (float)rows;
    float mu = sh[0] * inv;
    float var = sh2[0] * inv - mu * mu;
    mean[c] = mu;
    rstd[c] = rsqrtf(var + BN_EPS);
  }
}

// ---------------------------------------------------------------------------
// K3: h = relu(BN(y))  -> bf16 (WMMA operand for the node GEMMs)
// ---------------------------------------------------------------------------
__global__ void k_bn_relu(const float* __restrict__ y,
                          const float* __restrict__ mean,
                          const float* __restrict__ rstd,
                          const float* __restrict__ gamma,
                          const float* __restrict__ beta,
                          bf16_t* __restrict__ hbf) {
  int tid = blockIdx.x * blockDim.x + threadIdx.x;
  if (tid >= NN * HH) return;
  int c = tid & 127;
  float v = y[tid];
  v = gamma[c] * (v - mean[c]) * rstd[c] + beta[c];
  hbf[tid] = (bf16_t)fmaxf(v, 0.f);
}

// ---------------------------------------------------------------------------
// K4: a = h @ (W1-W2), b = h @ W2   (both [N,128]@[128,128], WMMA, A reused)
// ---------------------------------------------------------------------------
__global__ void k_gemm_node(const bf16_t* __restrict__ hbf,
                            const bf16_t* __restrict__ wA_t,
                            const bf16_t* __restrict__ wB_t,
                            float* __restrict__ a_n,
                            float* __restrict__ b_n) {
  const int wave = threadIdx.x >> 5;
  const int lane = threadIdx.x & 31;
  const int tile = blockIdx.x * 8 + wave;
  if (tile >= ROW_TILES) return;
  const int r = lane & 15, hi = lane >> 4;
  const int row0 = tile * 16;
  const bf16_t* arow = hbf + (size_t)(row0 + r) * HH;
  v16bf afr[4];
#pragma unroll
  for (int kb = 0; kb < 4; ++kb) afr[kb] = load_a_bf16(arow, 32 * kb, hi);
#pragma unroll
  for (int nt = 0; nt < 8; ++nt) {
    const int n0 = nt * 16;
    v8f accA = {}, accB = {};
#pragma unroll
    for (int kb = 0; kb < 4; ++kb) {
      v16bf bA = load_b_t(wA_t, HH, n0, 32 * kb, r, hi);
      accA = wmma_bf16(afr[kb], bA, accA);
      v16bf bB = load_b_t(wB_t, HH, n0, 32 * kb, r, hi);
      accB = wmma_bf16(afr[kb], bB, accB);
    }
    float* pa = a_n + (size_t)row0 * HH + n0 + r;
    float* pb = b_n + (size_t)row0 * HH + n0 + r;
#pragma unroll
    for (int rr = 0; rr < 8; ++rr) {
      pa[(size_t)(rr + 8 * hi) * HH] = accA[rr];
      pb[(size_t)(rr + 8 * hi) * HH] = accB[rr];
    }
  }
}

// ---------------------------------------------------------------------------
// K5: zero a buffer
// ---------------------------------------------------------------------------
__global__ void k_zero(float* __restrict__ p, int n) {
  int tid = blockIdx.x * blockDim.x + threadIdx.x;
  if (tid < n) p[tid] = 0.f;
}

// ---------------------------------------------------------------------------
// K6: edge stage: agg[src] += relu(a[src] + b[dst] + bias)
// 1 wave per edge; lane covers 4 features (float4). All arrays L2-resident.
// ---------------------------------------------------------------------------
__global__ void k_edge(const int* __restrict__ src, const int* __restrict__ dst,
                       const float* __restrict__ a_n, const float* __restrict__ b_n,
                       const float* __restrict__ bias, float* __restrict__ agg) {
  int gid = blockIdx.x * blockDim.x + threadIdx.x;
  int e = gid >> 5;
  if (e >= EE) return;
  int f = (gid & 31) * 4;
  int s = src[e], d = dst[e];
  v4f va = *(const v4f*)(a_n + (size_t)s * HH + f);
  v4f vb = *(const v4f*)(b_n + (size_t)d * HH + f);
  v4f vc = *(const v4f*)(bias + f);
  float* o = agg + (size_t)s * HH + f;
#pragma unroll
  for (int j = 0; j < 4; ++j) {
    float m = fmaxf(va[j] + vb[j] + vc[j], 0.f);
    atomicAdd(o + j, m);
  }
}

// ---------------------------------------------------------------------------
// K7: global sum-pool per graph: pool[seg[n]] += agg[n]
// ---------------------------------------------------------------------------
__global__ void k_pool(const int* __restrict__ seg, const float* __restrict__ agg,
                       float* __restrict__ pool) {
  int gid = blockIdx.x * blockDim.x + threadIdx.x;
  int n = gid >> 5;
  if (n >= NN) return;
  int f = (gid & 31) * 4;
  int g = seg[n];
  v4f v = *(const v4f*)(agg + (size_t)n * HH + f);
  float* o = pool + (size_t)g * HH + f;
#pragma unroll
  for (int j = 0; j < 4; ++j) atomicAdd(o + j, v[j]);
}

// ---------------------------------------------------------------------------
// K8: q = pool @ w_post + b_post   ([512,128]@[128,128], WMMA)
// single block, 8 waves x 4 row tiles = 32 tiles
// ---------------------------------------------------------------------------
__global__ void k_gemm_post(const float* __restrict__ pool,
                            const bf16_t* __restrict__ wpost_t,
                            const float* __restrict__ bias,
                            float* __restrict__ q) {
  const int wave = threadIdx.x >> 5;
  const int lane = threadIdx.x & 31;
  const int r = lane & 15, hi = lane >> 4;
  for (int t = wave * 4; t < wave * 4 + 4; ++t) {
    const int row0 = t * 16;
    const float* arow = pool + (size_t)(row0 + r) * HH;
    v16bf afr[4];
#pragma unroll
    for (int kb = 0; kb < 4; ++kb) afr[kb] = load_a_f32(arow, 32 * kb, hi);
#pragma unroll
    for (int nt = 0; nt < 8; ++nt) {
      const int n0 = nt * 16;
      v8f acc = {};
#pragma unroll
      for (int kb = 0; kb < 4; ++kb) {
        v16bf b = load_b_t(wpost_t, HH, n0, 32 * kb, r, hi);
        acc = wmma_bf16(afr[kb], b, acc);
      }
      const float bv = bias[n0 + r];
      float* qb = q + (size_t)row0 * HH + n0 + r;
#pragma unroll
      for (int rr = 0; rr < 8; ++rr)
        qb[(size_t)(rr + 8 * hi) * HH] = acc[rr] + bv;
    }
  }
}

// ---------------------------------------------------------------------------
// K9: out[g] = sigmoid( relu(BN(q[g])) . w_out + b_out )
// ---------------------------------------------------------------------------
__global__ void k_final(const float* __restrict__ q,
                        const float* __restrict__ mean, const float* __restrict__ rstd,
                        const float* __restrict__ gamma, const float* __restrict__ beta,
                        const float* __restrict__ w_out, const float* __restrict__ b_out,
                        float* __restrict__ out) {
  const int g = blockIdx.x;
  const int c = threadIdx.x;  // 128 threads
  float v = q[(size_t)g * HH + c];
  v = fmaxf(gamma[c] * (v - mean[c]) * rstd[c] + beta[c], 0.f);
  __shared__ float sh[128];
  sh[c] = v * w_out[c];
  __syncthreads();
  for (int o = 64; o > 0; o >>= 1) {
    if (c < o) sh[c] += sh[c + o];
    __syncthreads();
  }
  if (c == 0) {
    float z = sh[0] + b_out[0];
    out[g] = 1.f / (1.f + __expf(-z));
  }
}

// ---------------------------------------------------------------------------
// Launch
// ---------------------------------------------------------------------------
extern "C" void kernel_launch(void* const* d_in, const int* in_sizes, int n_in,
                              void* d_out, int out_size, void* d_ws, size_t ws_size,
                              hipStream_t stream) {
  const float* x      = (const float*)d_in[0];
  const int*   src    = (const int*)d_in[1];
  const int*   dst    = (const int*)d_in[2];
  const int*   seg    = (const int*)d_in[3];
  const float* w_pre  = (const float*)d_in[4];
  const float* b_pre  = (const float*)d_in[5];
  const float* g_pre  = (const float*)d_in[6];
  const float* be_pre = (const float*)d_in[7];
  const float* w_cn   = (const float*)d_in[8];
  const float* b_cn   = (const float*)d_in[9];
  const float* w_post = (const float*)d_in[10];
  const float* b_post = (const float*)d_in[11];
  const float* g_post = (const float*)d_in[12];
  const float* be_post= (const float*)d_in[13];
  const float* w_out  = (const float*)d_in[14];
  const float* b_out  = (const float*)d_in[15];
  float* out = (float*)d_out;

  char* ws = (char*)d_ws;
  size_t off = 0;
  auto take = [&](size_t bytes) -> void* {
    void* p = ws + off;
    off = (off + bytes + 255) & ~(size_t)255;
    return p;
  };
  // y doubles as the edge aggregation buffer after BN consumed it
  float*  y_agg   = (float*)take((size_t)NN * HH * 4);
  bf16_t* hbf     = (bf16_t*)take((size_t)NN * HH * 2);
  float*  a_n     = (float*)take((size_t)NN * HH * 4);
  float*  b_n     = (float*)take((size_t)NN * HH * 4);
  float*  pool    = (float*)take((size_t)GG * HH * 4);
  float*  q       = (float*)take((size_t)GG * HH * 4);
  float*  mean_pre  = (float*)take(HH * 4);
  float*  rstd_pre  = (float*)take(HH * 4);
  float*  mean_post = (float*)take(HH * 4);
  float*  rstd_post = (float*)take(HH * 4);
  bf16_t* wpre_t  = (bf16_t*)take((size_t)HH * FF * 2);
  bf16_t* wA_t    = (bf16_t*)take((size_t)HH * HH * 2);
  bf16_t* wB_t    = (bf16_t*)take((size_t)HH * HH * 2);
  bf16_t* wpost_t = (bf16_t*)take((size_t)HH * HH * 2);

  const int gemm_blocks = (ROW_TILES + 7) / 8;  // 391

  k_prep_w<<<(HH * HH + 255) / 256, 256, 0, stream>>>(w_pre, w_cn, w_post,
                                                      wpre_t, wA_t, wB_t, wpost_t);
  k_gemm_pre<<<gemm_blocks, 256, 0, stream>>>(x, wpre_t, b_pre, y_agg);
  k_colstats<<<HH, 256, 0, stream>>>(y_agg, NN, mean_pre, rstd_pre);
  k_bn_relu<<<(NN * HH + 255) / 256, 256, 0, stream>>>(y_agg, mean_pre, rstd_pre,
                                                       g_pre, be_pre, hbf);
  k_gemm_node<<<gemm_blocks, 256, 0, stream>>>(hbf, wA_t, wB_t, a_n, b_n);
  k_zero<<<(NN * HH + 255) / 256, 256, 0, stream>>>(y_agg, NN * HH);
  k_zero<<<(GG * HH + 255) / 256, 256, 0, stream>>>(pool, GG * HH);
  k_edge<<<(EE * 32) / 256, 256, 0, stream>>>(src, dst, a_n, b_n, b_cn, y_agg);
  k_pool<<<(NN * 32 + 255) / 256, 256, 0, stream>>>(seg, y_agg, pool);
  k_gemm_post<<<1, 256, 0, stream>>>(pool, wpost_t, b_post, q);
  k_colstats<<<HH, 256, 0, stream>>>(q, GG, mean_post, rstd_post);
  k_final<<<GG, HH, 0, stream>>>(q, mean_post, rstd_post, g_post, be_post,
                                 w_out, b_out, out);
}
